// S4Convolutional_20401094656325
// MI455X (gfx1250) — matrix operand order repaired and checked by reading, not compiled, not run on hardware
//
#include <hip/hip_runtime.h>
#include <hip/hip_bf16.h>

typedef __attribute__((ext_vector_type(16))) _Float16 v16h;
typedef __attribute__((ext_vector_type(8)))  _Float16 v8h;
typedef __attribute__((ext_vector_type(8)))  float    v8f;

#define LSEQ 2048
#define HDIM 512
#define BT 128            // t rows per block
#define BH 128            // h cols per block
#define KS 64             // s (K-dim) per staging step (2 WMMA sub-steps)
#define UP2 72            // LDS pitch in halves: 64 + 8 pad (144B rows)
#define UBUF (128 * UP2)  // one u-tile buffer (halves)
#define GRLEN (LSEQ + 80) // K[j] plus zero guard (max index 2109)

// -------- HiPPO matrix, closed form (reference's A = -(tril(PP^T)-diag(arange)))
__device__ __forceinline__ float hippoA(int i, int j) {
  if (j > i) return 0.0f;
  if (i == j) return -(float)(i + 1);
  return -__builtin_sqrtf((float)(2 * i + 1) * (float)(2 * j + 1));
}

// 64x64 f32 matmul in LDS, 1024 threads, syncs inside. d must differ from a,b.
__device__ __forceinline__ void mm64(float* __restrict__ d, const float* a,
                                     const float* b, int tid) {
  __syncthreads();
#pragma unroll
  for (int e = 0; e < 4; ++e) {
    int idx = tid + e * 1024;
    int r = idx >> 6, c = idx & 63;
    float s = 0.0f;
#pragma unroll 8
    for (int k = 0; k < 64; ++k) s = __builtin_fmaf(a[r * 64 + k], b[k * 64 + c], s);
    d[idx] = s;
  }
  __syncthreads();
}

// ---------------------------------------------------------------------------
// Kernel 1: compute K[l] = C*A_bar^l*B_bar  (single workgroup, 1024 threads)
// ---------------------------------------------------------------------------
__global__ __launch_bounds__(1024) void s4_setup(const float* __restrict__ B_ssm,
                                                 const float* __restrict__ C_ssm,
                                                 const float* __restrict__ log_dt,
                                                 float* __restrict__ g) {
  __shared__ float T[4096];      // scaled dt*A / scratch
  __shared__ float P[4096];      // Horner acc -> A_bar -> M256
  __shared__ float Q[4096];      // scratch
  __shared__ float Wl[16 * 64];  // rows C*A_bar^j
  __shared__ float Vl[2][64 * 16];
  __shared__ float rowsum[64], rhs[64], xs[64];
  __shared__ float sh_scl;
  __shared__ int sh_nsq;
  const int tid = threadIdx.x;

  if (tid < 64) {
    float s = 0.0f;
    for (int j = 0; j <= tid; ++j) s += __builtin_fabsf(hippoA(tid, j));
    rowsum[tid] = s;
  }
  __syncthreads();
  if (tid == 0) {
    float dt = __expf(log_dt[0]);
    float nrm = 0.0f;
    for (int i = 0; i < 64; ++i) nrm = fmaxf(nrm, rowsum[i]);
    nrm *= dt;
    int s = 0;
    while (nrm > 0.25f && s < 30) { nrm *= 0.5f; ++s; }
    sh_scl = dt * exp2f(-(float)s);
    sh_nsq = s;
  }
  __syncthreads();
  const float scl = sh_scl;
  const int nsq = sh_nsq;

  // T = scl*A ; P = I
#pragma unroll
  for (int e = 0; e < 4; ++e) {
    int idx = tid + e * 1024;
    int r = idx >> 6, c = idx & 63;
    T[idx] = scl * hippoA(r, c);
    P[idx] = (r == c) ? 1.0f : 0.0f;
  }
  __syncthreads();
  // Horner Taylor degree 10: P = I + T*P/k
  for (int k = 10; k >= 1; --k) {
    mm64(Q, T, P, tid);
    float invk = 1.0f / (float)k;
#pragma unroll
    for (int e = 0; e < 4; ++e) {
      int idx = tid + e * 1024;
      int r = idx >> 6, c = idx & 63;
      P[idx] = Q[idx] * invk + ((r == c) ? 1.0f : 0.0f);
    }
    __syncthreads();
  }
  // squarings
  for (int q = 0; q < nsq; ++q) {
    mm64(Q, P, P, tid);
#pragma unroll
    for (int e = 0; e < 4; ++e) { int idx = tid + e * 1024; P[idx] = Q[idx]; }
    __syncthreads();
  }
  // P = A_bar.  rhs = (A_bar - I) @ B_ssm
  if (tid < 64) {
    float s = 0.0f;
    for (int j = 0; j < 64; ++j)
      s += (P[tid * 64 + j] - ((tid == j) ? 1.0f : 0.0f)) * B_ssm[j];
    rhs[tid] = s;
  }
  __syncthreads();
  if (tid == 0) {  // forward substitution: A x = rhs (A lower triangular)
    for (int i = 0; i < 64; ++i) {
      float s = rhs[i];
      for (int j = 0; j < i; ++j) s -= hippoA(i, j) * xs[j];
      xs[i] = s / (-(float)(i + 1));
    }
  }
  __syncthreads();
  // W rows: W[0]=C ; W[j] = W[j-1] @ A_bar
  if (tid < 64) Wl[tid] = C_ssm[tid];
  __syncthreads();
  for (int j = 1; j < 16; ++j) {
    if (tid < 64) {
      float s = 0.0f;
      for (int k = 0; k < 64; ++k) s += Wl[(j - 1) * 64 + k] * P[k * 64 + tid];
      Wl[j * 64 + tid] = s;
    }
    __syncthreads();
  }
  // M16 = A_bar^16  (ends in T)
  mm64(Q, P, P, tid);
  mm64(T, Q, Q, tid);
  mm64(Q, T, T, tid);
  mm64(T, Q, Q, tid);
  // V columns: c_0 = B_bar ; c_i = M16 @ c_{i-1}
  if (tid < 64) Vl[0][tid * 16 + 0] = xs[tid];
  __syncthreads();
  for (int i = 1; i < 16; ++i) {
    if (tid < 64) {
      float s = 0.0f;
      for (int k = 0; k < 64; ++k) s += T[tid * 64 + k] * Vl[0][k * 16 + (i - 1)];
      Vl[0][tid * 16 + i] = s;
    }
    __syncthreads();
  }
  // M256 = M16^16 (ends in P; A_bar no longer needed)
  mm64(Q, T, T, tid);
  mm64(P, Q, Q, tid);
  mm64(Q, P, P, tid);
  mm64(P, Q, Q, tid);
  // 8 chunks: K[256q+16i+j] = (W @ V)[j][i] ; V <- M256 @ V
  int cur = 0;
  for (int q = 0; q < 8; ++q) {
    if (tid < 256) {
      int j = tid & 15, i = tid >> 4;
      float s = 0.0f;
      for (int k = 0; k < 64; ++k) s += Wl[j * 64 + k] * Vl[cur][k * 16 + i];
      g[256 * q + 16 * i + j] = s;  // store K un-reversed: g[l] = K[l]
    }
    __syncthreads();
    if (q < 7) {
      {
        int r = tid >> 4, i = tid & 15;
        float s = 0.0f;
        for (int k = 0; k < 64; ++k) s += P[r * 64 + k] * Vl[cur][k * 16 + i];
        Vl[cur ^ 1][tid] = s;
      }
      __syncthreads();
      cur ^= 1;
    }
  }
}

// ---------------------------------------------------------------------------
// Kernel 2: y[b,t,h] = sum_{s<=t} K[2047-t+s]*u[b,s,h] + D*u[b,t,h]
//   WMMA mapping: D[h,t] = A(u^T tile, 16h x 32s) x B(Toeplitz, 32s x 16t)
//   B gather is ascending-contiguous in K[] -> pure ds_load_b128, no reversal.
// ---------------------------------------------------------------------------
__global__ __launch_bounds__(256) void s4_conv(const float* __restrict__ u,
                                               const float* __restrict__ gws,
                                               const float* __restrict__ Dp,
                                               float* __restrict__ y) {
  __shared__ __align__(16) _Float16 gr[GRLEN];    // gr[j]=K[j], j>=2048 -> 0
  __shared__ __align__(16) _Float16 ut[2 * UBUF]; // double-buffered [h][s] tile

  const int tid = threadIdx.x;
  const int b = blockIdx.x, tblk = blockIdx.y, hblk = blockIdx.z;
  const int t0 = tblk * BT, h0 = hblk * BH;
  const int wave = tid >> 5, lane = tid & 31;
  const int wh2 = wave >> 2;  // h half (0..1), 64 h per wave
  const int wtt = wave & 3;   // t quarter (0..3), 32 t per wave
  const int laneHi = lane >> 4, lane15 = lane & 15;

  // stage K (zero guard above) once
  for (int i = tid; i < GRLEN; i += 256)
    gr[i] = (_Float16)((i < LSEQ) ? gws[i] : 0.0f);

  v8f acc[4][2];
#pragma unroll
  for (int m = 0; m < 4; ++m)
#pragma unroll
    for (int n = 0; n < 2; ++n)
#pragma unroll
      for (int r = 0; r < 8; ++r) acc[m][n][r] = 0.0f;

  const float* ub = u + (size_t)b * LSEQ * HDIM;
  const int hrow = tid >> 1;        // staging: h row handled by this thread
  const int sblk = (tid & 1) * 32;  // staging: s chunk (0 or 32)
  const int tmax = t0 + wtt * 32 + 31;
  const int ksteps = (t0 + BT) / KS;  // causal bound: s <= t0+BT-1

  // prologue: stage first tile into buffer 0
  {
    const float* src = &ub[(size_t)sblk * HDIM + h0 + hrow];
    _Float16* dst = &ut[hrow * UP2 + sblk];
#pragma unroll
    for (int c = 0; c < 4; ++c) {
      v8h pk;
#pragma unroll
      for (int j = 0; j < 8; ++j) pk[j] = (_Float16)src[(size_t)(c * 8 + j) * HDIM];
      *(v8h*)(dst + c * 8) = pk;
    }
  }
  __syncthreads();

  for (int kb = 0; kb < ksteps; ++kb) {
    const int s0 = kb * KS;
    const int cur = kb & 1;
    const bool havenext = (kb + 1 < ksteps);

    // issue next tile's global loads early (overlap with WMMA below)
    float stg[32];
    if (havenext) {
      const float* src = &ub[(size_t)(s0 + KS + sblk) * HDIM + h0 + hrow];
#pragma unroll
      for (int j = 0; j < 32; ++j) stg[j] = src[(size_t)j * HDIM];
    }

    const _Float16* utc = &ut[cur * UBUF];
#pragma unroll
    for (int half = 0; half < 2; ++half) {
      const int ssub = s0 + half * 32;
      if (ssub <= tmax) {  // wave-uniform diagonal skip (EXEC stays all-1)
        // A operands (u): contiguous 16B-aligned runs -> 2x ds_load_b128 each
        v16h aop[4];
#pragma unroll
        for (int mh = 0; mh < 4; ++mh) {
          const _Float16* rp =
              &utc[(wh2 * 64 + mh * 16 + lane15) * UP2 + half * 32 + laneHi * 16];
          v8h lo = *(const v8h*)(rp);
          v8h hi = *(const v8h*)(rp + 8);
          aop[mh] = __builtin_shufflevector(lo, hi, 0, 1, 2, 3, 4, 5, 6, 7, 8, 9,
                                            10, 11, 12, 13, 14, 15);
        }
        // B operands (Toeplitz): bop[p] = K[2047 - t + s], ascending contiguous
#pragma unroll
        for (int nt = 0; nt < 2; ++nt) {
          const int t = t0 + wtt * 32 + nt * 16 + lane15;
          const int baseR = (LSEQ - 1) - t + ssub + laneHi * 16;
          v16h bop;
#pragma unroll
          for (int p = 0; p < 16; ++p) bop[p] = gr[baseR + p];
#pragma unroll
          for (int mh = 0; mh < 4; ++mh)
            acc[mh][nt] = __builtin_amdgcn_wmma_f32_16x16x32_f16(
                false, aop[mh], false, bop, (short)0, acc[mh][nt], false, false);
        }
      }
    }

    // convert + store next tile into the other buffer
    if (havenext) {
      _Float16* dst = &ut[(cur ^ 1) * UBUF + hrow * UP2 + sblk];
#pragma unroll
      for (int c = 0; c < 4; ++c) {
        v8h pk;
#pragma unroll
        for (int j = 0; j < 8; ++j) pk[j] = (_Float16)stg[c * 8 + j];
        *(v8h*)(dst + c * 8) = pk;
      }
    }
    __syncthreads();
  }

  // epilogue: y = acc + D*u   (h is fastest axis -> float4 global ops)
  const float D = Dp[0];
#pragma unroll
  for (int mh = 0; mh < 4; ++mh)
#pragma unroll
    for (int nt = 0; nt < 2; ++nt) {
      const int t = t0 + wtt * 32 + nt * 16 + lane15;
      const int h = h0 + wh2 * 64 + mh * 16 + laneHi * 8;  // + r (r=0..7)
      const size_t idx = (size_t)b * LSEQ * HDIM + (size_t)t * HDIM + h;
      float4 u0 = *(const float4*)&u[idx];
      float4 u1 = *(const float4*)&u[idx + 4];
      float4 o0, o1;
      o0.x = acc[mh][nt][0] + D * u0.x;
      o0.y = acc[mh][nt][1] + D * u0.y;
      o0.z = acc[mh][nt][2] + D * u0.z;
      o0.w = acc[mh][nt][3] + D * u0.w;
      o1.x = acc[mh][nt][4] + D * u1.x;
      o1.y = acc[mh][nt][5] + D * u1.y;
      o1.z = acc[mh][nt][6] + D * u1.z;
      o1.w = acc[mh][nt][7] + D * u1.w;
      *(float4*)&y[idx] = o0;
      *(float4*)&y[idx + 4] = o1;
    }
}

extern "C" void kernel_launch(void* const* d_in, const int* in_sizes, int n_in,
                              void* d_out, int out_size, void* d_ws, size_t ws_size,
                              hipStream_t stream) {
  const float* u      = (const float*)d_in[0];
  const float* B_ssm  = (const float*)d_in[1];
  const float* C_ssm  = (const float*)d_in[2];
  const float* D_skip = (const float*)d_in[3];
  const float* log_dt = (const float*)d_in[4];
  float* g = (float*)d_ws;  // 2048 f32: K[l]

  s4_setup<<<1, 1024, 0, stream>>>(B_ssm, C_ssm, log_dt, g);

  dim3 grid(16, LSEQ / BT, HDIM / BH);  // (b, t-block, h-block) = (16,16,4)
  s4_conv<<<grid, 256, 0, stream>>>(u, g, D_skip, (float*)d_out);
}
